// GPTModel_16741782520524
// MI455X (gfx1250) — matrix operand order repaired and checked by reading, not compiled, not run on hardware
//
#include <hip/hip_runtime.h>

// ---------------------------------------------------------------------------
// GPT forward (6 layers, B=4 T=1024 D=768 H=12 HD=64 FF=3072 V=32000) for
// MI455X (gfx1250, wave32). All dense GEMMs run on v_wmma_f32_16x16x32_f16.
// Roofline: ~627 GFLOP total; ~1 GB of logits traffic (write + loss re-read)
// dominates at 23.3 TB/s -> HBM-bound. Each wave computes a 64x16 tile
// (4 accumulators) so every packed B fragment feeds 4 WMMAs: 4x the FLOP per
// weight byte, and the 4 independent accumulators cover the WMMA->WMMA RAW
// hazard distance without NOPs.
// ---------------------------------------------------------------------------

typedef __attribute__((ext_vector_type(16))) _Float16 v16h;
typedef __attribute__((ext_vector_type(8)))  _Float16 v8h;
typedef __attribute__((ext_vector_type(8)))  float    v8f;

#define NB  4
#define NT  1024
#define ND  768
#define NH  12
#define NHD 64
#define NL  6
#define NFF 3072
#define NV  32000
#define NTOK (NB * NT)   // 4096 rows

// ---------------------------------------------------------------------------
// WMMA GEMM: C[M,N] = act(A[M,K] @ Bpacked + bias [+ X residual])
// A: row-major f16. Bp: fragment-packed f16 (see pack kernels).
// 128 threads = 4 waves; wave w computes a 64x16 tile at
// (64*blockIdx.x .. +63, 16*(4*blockIdx.y + w) .. +15).
// ---------------------------------------------------------------------------
template <bool RELU, bool RESID, bool BIAS, bool OUT32, bool OUT16>
__global__ void wmma_gemm_kernel(const _Float16* __restrict__ A,
                                 const _Float16* __restrict__ Bp,
                                 const float* __restrict__ bias,
                                 const float* __restrict__ X,
                                 float* __restrict__ C32,
                                 _Float16* __restrict__ C16,
                                 int M, int N, int K) {
    const int lane   = threadIdx.x & 31;
    const int wave   = threadIdx.x >> 5;
    const int mBase0 = blockIdx.x << 6;               // 64-row group
    const int nTile  = blockIdx.y * 4 + wave;
    const int nBase  = nTile << 4;
    const int l  = lane & 15;
    const int hi = lane >> 4;
    const int nTiles = N >> 4;

    // A fragment source: lane l<16 holds row mBase+l, K chunks [k..k+7],[k+16..k+23]
    const _Float16* Arow = A + (size_t)(mBase0 + l) * K + hi * 8;

    v8f acc[4] = {};
    for (int kt = 0; kt < K; kt += 32) {
        const size_t fid = (size_t)(kt >> 5) * nTiles + nTile;
        const _Float16* bptr = Bp + (fid * 32 + lane) * 16;
        v16h b = *(const v16h*)bptr;
        if (kt + 32 < K)  // prefetch next B fragment (global_prefetch_b8)
            __builtin_prefetch(bptr + (size_t)nTiles * 512, 0, 1);

#pragma unroll
        for (int j = 0; j < 4; ++j) {
            const _Float16* aj = Arow + (size_t)(j * 16) * K + kt;
            v8h a0 = *(const v8h*)(aj);
            v8h a1 = *(const v8h*)(aj + 16);
            v16h a;
#pragma unroll
            for (int i = 0; i < 8; ++i) { a[i] = a0[i]; a[i + 8] = a1[i]; }
            acc[j] = __builtin_amdgcn_wmma_f32_16x16x32_f16(
                false, a, false, b, (short)0, acc[j], false, false);
        }
    }

    // Epilogue: acc[j][r] -> C[mBase0 + 16*j + 8*hi + r][nBase + l]
    const int n = nBase + l;
    const float bn = BIAS ? bias[n] : 0.0f;
#pragma unroll
    for (int j = 0; j < 4; ++j) {
#pragma unroll
        for (int r = 0; r < 8; ++r) {
            const int m = mBase0 + j * 16 + hi * 8 + r;
            float v = acc[j][r] + bn;
            if (RESID) v += X[(size_t)m * N + n];
            if (RELU)  v = v > 0.0f ? v : 0.0f;
            if (OUT32) C32[(size_t)m * N + n] = v;
            if (OUT16) C16[(size_t)m * N + n] = (_Float16)v;
        }
    }
}

// ---------------------------------------------------------------------------
// Weight packing into WMMA B-fragment order (f32 -> f16).
// B logical [K,N]; dest half index: fid=(k/32)*(N/16)+(n/16),
// lane = 16*((k%32/8)&1) + n%16, i = k%8 + (k%32>=16 ? 8 : 0).
// ---------------------------------------------------------------------------
__device__ __forceinline__ void pack_store(_Float16* dst, int k, int n, int N, float v) {
    const int km   = k & 31;
    const int lane = (((km >> 3) & 1) << 4) + (n & 15);
    const int i    = (km & 7) + ((km >> 4) << 3);
    const size_t fid = (size_t)(k >> 5) * (N >> 4) + (n >> 4);
    dst[(fid * 32 + lane) * 16 + i] = (_Float16)v;
}

__global__ void pack_b_rowmajor_kernel(const float* __restrict__ src,
                                       _Float16* __restrict__ dst, int K, int N) {
    size_t idx = blockIdx.x * (size_t)blockDim.x + threadIdx.x;
    if (idx >= (size_t)K * N) return;
    int k = (int)(idx / N), n = (int)(idx % N);
    pack_store(dst, k, n, N, src[idx]);
}

// W shaped (H, D, HD) -> logical B[d][h*HD+e]
__global__ void pack_b_headed_kernel(const float* __restrict__ src,
                                     _Float16* __restrict__ dst, int D, int H, int HD) {
    const int N = H * HD;
    size_t idx = blockIdx.x * (size_t)blockDim.x + threadIdx.x;
    if (idx >= (size_t)D * N) return;
    int k = (int)(idx / N), n = (int)(idx % N);
    int h = n / HD, e = n % HD;
    pack_store(dst, k, n, N, src[((size_t)h * D + k) * HD + e]);
}

// ---------------------------------------------------------------------------
// Embedding: x[b,t,:] = tok_emb[idx[b,t],:] + pos_emb[t,:]
// ---------------------------------------------------------------------------
__global__ void embed_kernel(const int* __restrict__ tok,
                             const float* __restrict__ te,
                             const float* __restrict__ pe,
                             float* __restrict__ x) {
    size_t idx = blockIdx.x * (size_t)blockDim.x + threadIdx.x;
    if (idx >= (size_t)NTOK * ND) return;
    int row = (int)(idx / ND), d = (int)(idx % ND);
    int t = row % NT;
    x[idx] = te[(size_t)tok[row] * ND + d] + pe[(size_t)t * ND + d];
}

// ---------------------------------------------------------------------------
// LayerNorm (biased var, eps 1e-5), f32 in -> f16 out. One block per row.
// ---------------------------------------------------------------------------
__global__ void layernorm_f16_kernel(const float* __restrict__ x,
                                     const float* __restrict__ g,
                                     const float* __restrict__ b,
                                     _Float16* __restrict__ out, int D) {
    __shared__ float red[256];
    const int row = blockIdx.x, tid = threadIdx.x;
    const float* xr = x + (size_t)row * D;

    float s = 0.0f;
    for (int d = tid; d < D; d += 256) s += xr[d];
    red[tid] = s; __syncthreads();
    for (int off = 128; off > 0; off >>= 1) {
        if (tid < off) red[tid] += red[tid + off];
        __syncthreads();
    }
    const float mean = red[0] / (float)D;
    __syncthreads();

    float v = 0.0f;
    for (int d = tid; d < D; d += 256) { float t = xr[d] - mean; v += t * t; }
    red[tid] = v; __syncthreads();
    for (int off = 128; off > 0; off >>= 1) {
        if (tid < off) red[tid] += red[tid + off];
        __syncthreads();
    }
    const float rstd = rsqrtf(red[0] / (float)D + 1e-5f);

    for (int d = tid; d < D; d += 256)
        out[(size_t)row * D + d] = (_Float16)((xr[d] - mean) * rstd * g[d] + b[d]);
}

// ---------------------------------------------------------------------------
// Causal attention, one block (128 thr) per (b,h,t) query row.
// scores[s] = scale * dot(k[t], q[s]) for s<=t (ref: einsum('bhte,bhse->bhts')),
// softmax over s, o[t,e] = sum_s w[s] * v[s,e]. Scores staged in LDS (4KB).
// ---------------------------------------------------------------------------
__global__ void attention_kernel(const _Float16* __restrict__ q,
                                 const _Float16* __restrict__ k,
                                 const _Float16* __restrict__ v,
                                 _Float16* __restrict__ o, float scale) {
    __shared__ float sc[NT];
    __shared__ float kq[NHD];
    __shared__ float red[128];
    const int t = blockIdx.x, h = blockIdx.y, b = blockIdx.z;
    const int tid = threadIdx.x;
    const size_t rowBase = (size_t)b * NT * ND + (size_t)h * NHD;

    for (int e = tid; e < NHD; e += 128)
        kq[e] = (float)k[rowBase + (size_t)t * ND + e];
    __syncthreads();

    for (int s = tid; s <= t; s += 128) {
        const _Float16* qs = q + rowBase + (size_t)s * ND;
        float d = 0.0f;
#pragma unroll
        for (int e = 0; e < NHD; ++e) d += kq[e] * (float)qs[e];
        sc[s] = d * scale;
    }
    __syncthreads();

    float m = -3.4e38f;
    for (int s = tid; s <= t; s += 128) m = fmaxf(m, sc[s]);
    red[tid] = m; __syncthreads();
    for (int off = 64; off > 0; off >>= 1) {
        if (tid < off) red[tid] = fmaxf(red[tid], red[tid + off]);
        __syncthreads();
    }
    m = red[0];
    __syncthreads();

    float z = 0.0f;
    for (int s = tid; s <= t; s += 128) {
        float p = __expf(sc[s] - m);
        sc[s] = p;
        z += p;
    }
    red[tid] = z; __syncthreads();
    for (int off = 64; off > 0; off >>= 1) {
        if (tid < off) red[tid] += red[tid + off];
        __syncthreads();
    }
    const float invZ = 1.0f / red[0];
    __syncthreads();

    const int e = tid & 63, halfi = tid >> 6;
    float acc = 0.0f;
    for (int s = halfi; s <= t; s += 2)
        acc += sc[s] * (float)v[rowBase + (size_t)s * ND + e];
    red[tid] = acc; __syncthreads();
    if (tid < 64)
        o[rowBase + (size_t)t * ND + e] = (_Float16)((red[tid] + red[tid + 64]) * invZ);
}

// ---------------------------------------------------------------------------
// f32 -> f16 elementwise (residual stream -> LM-head A operand)
// ---------------------------------------------------------------------------
__global__ void f32_to_f16_kernel(const float* __restrict__ src,
                                  _Float16* __restrict__ dst, size_t n) {
    size_t i = blockIdx.x * (size_t)blockDim.x + threadIdx.x;
    if (i < n) dst[i] = (_Float16)src[i];
}

// ---------------------------------------------------------------------------
// Loss: per-row NLL over V=32000, then deterministic single-block mean.
// ---------------------------------------------------------------------------
__global__ void row_nll_kernel(const float* __restrict__ logits,
                               const int* __restrict__ tgt,
                               float* __restrict__ nll) {
    __shared__ float red[256];
    const int row = blockIdx.x, tid = threadIdx.x;
    const float* lr = logits + (size_t)row * NV;

    float m = -3.4e38f;
    for (int i = tid; i < NV; i += 256) m = fmaxf(m, lr[i]);
    red[tid] = m; __syncthreads();
    for (int off = 128; off > 0; off >>= 1) {
        if (tid < off) red[tid] = fmaxf(red[tid], red[tid + off]);
        __syncthreads();
    }
    m = red[0];
    __syncthreads();

    float z = 0.0f;
    for (int i = tid; i < NV; i += 256) z += __expf(lr[i] - m);
    red[tid] = z; __syncthreads();
    for (int off = 128; off > 0; off >>= 1) {
        if (tid < off) red[tid] += red[tid + off];
        __syncthreads();
    }
    if (tid == 0) nll[row] = -(lr[tgt[row]] - m - __logf(red[0]));
}

__global__ void loss_reduce_kernel(const float* __restrict__ nll,
                                   float* __restrict__ out, int n) {
    __shared__ float red[256];
    const int tid = threadIdx.x;
    float s = 0.0f;
    for (int i = tid; i < n; i += 256) s += nll[i];
    red[tid] = s; __syncthreads();
    for (int off = 128; off > 0; off >>= 1) {
        if (tid < off) red[tid] += red[tid + off];
        __syncthreads();
    }
    if (tid == 0) out[0] = red[0] / (float)n;
}

// ---------------------------------------------------------------------------
// Host-side launch helpers
// ---------------------------------------------------------------------------
template <bool RELU, bool RESID, bool BIAS, bool OUT32, bool OUT16>
static void launch_gemm(const _Float16* A, const _Float16* Bp, const float* bias,
                        const float* X, float* C32, _Float16* C16,
                        int M, int N, int K, hipStream_t s) {
    dim3 grid(M / 64, (N / 16) / 4);   // wave = 64x16 tile, block = 64x64
    wmma_gemm_kernel<RELU, RESID, BIAS, OUT32, OUT16>
        <<<grid, dim3(128), 0, s>>>(A, Bp, bias, X, C32, C16, M, N, K);
}

static void launch_pack_rm(const float* src, _Float16* dst, int K, int N, hipStream_t s) {
    size_t n = (size_t)K * N;
    pack_b_rowmajor_kernel<<<dim3((unsigned)((n + 255) / 256)), dim3(256), 0, s>>>(src, dst, K, N);
}

extern "C" void kernel_launch(void* const* d_in, const int* in_sizes, int n_in,
                              void* d_out, int out_size, void* d_ws, size_t ws_size,
                              hipStream_t stream) {
    (void)in_sizes; (void)n_in; (void)out_size; (void)ws_size;
    // Inputs (setup_inputs order)
    const int*   tok     = (const int*)  d_in[0];
    const int*   targets = (const int*)  d_in[1];
    const float* tok_emb = (const float*)d_in[2];
    const float* pos_emb = (const float*)d_in[3];
    const float* Wk      = (const float*)d_in[4];
    const float* Wq      = (const float*)d_in[5];
    const float* Wv      = (const float*)d_in[6];
    const float* Wproj   = (const float*)d_in[7];
    const float* bproj   = (const float*)d_in[8];
    const float* ln1_g   = (const float*)d_in[9];
    const float* ln1_b   = (const float*)d_in[10];
    const float* ln2_g   = (const float*)d_in[11];
    const float* ln2_b   = (const float*)d_in[12];
    const float* W1      = (const float*)d_in[13];
    const float* b1      = (const float*)d_in[14];
    const float* W2      = (const float*)d_in[15];
    const float* b2      = (const float*)d_in[16];
    const float* lm_w    = (const float*)d_in[17];
    const float* lm_b    = (const float*)d_in[18];

    float* logits = (float*)d_out;                       // [4096, 32000]
    float* loss   = logits + (size_t)NTOK * NV;          // scalar

    // Workspace carve-out (256B aligned)
    char*  ws  = (char*)d_ws;
    size_t cur = 0;
    auto carve = [&](size_t bytes) {
        void* p = ws + cur;
        cur += (bytes + 255) & ~(size_t)255;
        return p;
    };
    float*    x     = (float*)   carve((size_t)NTOK * ND * 4);   // residual stream
    _Float16* h16   = (_Float16*)carve((size_t)NTOK * ND * 2);   // LN output
    _Float16* q16   = (_Float16*)carve((size_t)NTOK * ND * 2);
    _Float16* k16   = (_Float16*)carve((size_t)NTOK * ND * 2);
    _Float16* v16   = (_Float16*)carve((size_t)NTOK * ND * 2);
    _Float16* o16   = (_Float16*)carve((size_t)NTOK * ND * 2);
    _Float16* x16   = (_Float16*)carve((size_t)NTOK * ND * 2);
    _Float16* a16   = (_Float16*)carve((size_t)NTOK * NFF * 2);  // MLP hidden
    _Float16* wpack = (_Float16*)carve((size_t)ND * NV * 2);     // packed weights (reused)
    float*    nll   = (float*)   carve((size_t)NTOK * 4);

    const float scale = rsqrtf((float)ND);  // reference scales by d_model^-0.5

    // 1) Embedding
    {
        size_t n = (size_t)NTOK * ND;
        embed_kernel<<<dim3((unsigned)((n + 255) / 256)), dim3(256), 0, stream>>>(
            tok, tok_emb, pos_emb, x);
    }

    // 2) Transformer layers
    for (int l = 0; l < NL; ++l) {
        const size_t wqkvOff = (size_t)l * NH * ND * NHD;

        layernorm_f16_kernel<<<dim3(NTOK), dim3(256), 0, stream>>>(
            x, ln1_g + (size_t)l * ND, ln1_b + (size_t)l * ND, h16, ND);

        // Q, K, V projections (WMMA)
        {
            size_t n = (size_t)ND * ND;
            dim3 pg((unsigned)((n + 255) / 256)), pb(256);
            pack_b_headed_kernel<<<pg, pb, 0, stream>>>(Wq + wqkvOff, wpack, ND, NH, NHD);
            launch_gemm<false, false, false, false, true>(
                h16, wpack, nullptr, nullptr, nullptr, q16, NTOK, ND, ND, stream);
            pack_b_headed_kernel<<<pg, pb, 0, stream>>>(Wk + wqkvOff, wpack, ND, NH, NHD);
            launch_gemm<false, false, false, false, true>(
                h16, wpack, nullptr, nullptr, nullptr, k16, NTOK, ND, ND, stream);
            pack_b_headed_kernel<<<pg, pb, 0, stream>>>(Wv + wqkvOff, wpack, ND, NH, NHD);
            launch_gemm<false, false, false, false, true>(
                h16, wpack, nullptr, nullptr, nullptr, v16, NTOK, ND, ND, stream);
        }

        // Causal attention
        attention_kernel<<<dim3(NT, NH, NB), dim3(128), 0, stream>>>(
            q16, k16, v16, o16, scale);

        // Output projection + bias + residual -> x (WMMA)
        launch_pack_rm(Wproj + (size_t)l * ND * ND, wpack, ND, ND, stream);
        launch_gemm<false, true, true, true, false>(
            o16, wpack, bproj + (size_t)l * ND, x, x, nullptr, NTOK, ND, ND, stream);

        // MLP
        layernorm_f16_kernel<<<dim3(NTOK), dim3(256), 0, stream>>>(
            x, ln2_g + (size_t)l * ND, ln2_b + (size_t)l * ND, h16, ND);

        launch_pack_rm(W1 + (size_t)l * ND * NFF, wpack, ND, NFF, stream);
        launch_gemm<true, false, true, false, true>(   // ReLU(h @ W1 + b1) -> f16
            h16, wpack, b1 + (size_t)l * NFF, nullptr, nullptr, a16, NTOK, NFF, ND, stream);

        launch_pack_rm(W2 + (size_t)l * NFF * ND, wpack, NFF, ND, stream);
        launch_gemm<false, true, true, true, false>(   // x += a @ W2 + b2
            a16, wpack, b2 + (size_t)l * ND, x, x, nullptr, NTOK, ND, NFF, stream);
    }

    // 3) LM head (no final LN in reference): logits = x @ lm_w + lm_b
    {
        size_t n = (size_t)NTOK * ND;
        f32_to_f16_kernel<<<dim3((unsigned)((n + 255) / 256)), dim3(256), 0, stream>>>(
            x, x16, n);
    }
    launch_pack_rm(lm_w, wpack, ND, NV, stream);
    launch_gemm<false, false, true, true, false>(
        x16, wpack, lm_b, nullptr, logits, nullptr, NTOK, NV, ND, stream);

    // 4) Loss (deterministic: per-row NLL, then ordered single-block mean)
    row_nll_kernel<<<dim3(NTOK), dim3(256), 0, stream>>>(logits, targets, nll);
    loss_reduce_kernel<<<dim3(1), dim3(256), 0, stream>>>(nll, loss, NTOK);
}
